// Attention_46918222741521
// MI455X (gfx1250) — matrix operand order repaired and checked by reading, not compiled, not run on hardware
//
#include <hip/hip_runtime.h>
#include <hip/hip_bf16.h>

// SAN pairwise-attention block, fused for MI455X (gfx1250, wave32, WMMA).
// All matmul-shaped work goes through v_wmma_f32_16x16x32_f16.

typedef __attribute__((ext_vector_type(16))) _Float16 v16h;
typedef __attribute__((ext_vector_type(8)))  float    v8f;

#define N_   4
#define CIN  256
#define H_   56
#define W_   56
#define HW_  3136
#define REL  32
#define KS_  7
#define KK_  49
#define ROWS 320   // 32(q)+32(k)+256(v)

union Frag16 { v16h h; uint4 u[2]; };          // 32B: one f16 WMMA A/B fragment per lane
union Half8  { _Float16 h[8]; uint4 u; };      // 16B packed halves
union AccV   { v8f v; float4 f[2]; };          // C/D fragment per lane

__device__ __forceinline__ v8f wmma16(v16h a, v16h b, v8f c) {
  return __builtin_amdgcn_wmma_f32_16x16x32_f16(false, a, false, b, (short)0, c,
                                                false, false);
}

__device__ __forceinline__ int refl(int t, int n) {
  if (t < 0) t = -t;
  if (t >= n) t = 2 * n - 2 - t;
  return t;
}

// ---- prep: Wcat f16 [320][256] = w1|w2|w3 -----------------------------------
__global__ void prep_w(const float* __restrict__ w1, const float* __restrict__ w2,
                       const float* __restrict__ w3, _Float16* __restrict__ Wcat) {
  int i = blockIdx.x * 256 + threadIdx.x;
  if (i >= ROWS * CIN) return;
  int c = i % CIN, r = i / CIN;
  float v = (r < 32) ? w1[r * CIN + c]
          : (r < 64) ? w2[(r - 32) * CIN + c]
                     : w3[(r - 64) * CIN + c];
  Wcat[i] = (_Float16)v;
}

// ---- prep: x transpose -> pixel-major f16 [N][HW][256] ----------------------
__global__ void prep_xt(const float* __restrict__ x, _Float16* __restrict__ xt) {
  size_t i = (size_t)blockIdx.x * 256 + threadIdx.x;
  if (i >= (size_t)N_ * CIN * HW_) return;
  int pix = (int)(i % HW_);
  int ch  = (int)((i / HW_) % CIN);
  int n   = (int)(i / ((size_t)HW_ * CIN));
  xt[((size_t)n * HW_ + pix) * CIN + ch] = (_Float16)x[i];
}

// ---- kernel 1: qkv projection GEMM, WMMA f16 --------------------------------
// Out[320 x 3136] per batch = Wcat(320x256) * x_n(256x3136). 1 wave = 16x16 tile.
__global__ void k1_gemm(const _Float16* __restrict__ Wcat,
                        const _Float16* __restrict__ xt,
                        float* __restrict__ q, float* __restrict__ kb,
                        float* __restrict__ v) {
  const int lane = threadIdx.x & 31;
  const int wave = threadIdx.x >> 5;
  const int wid  = blockIdx.x * 4 + wave;          // 20 mtiles * 784 coltiles
  const int mtile = wid / 784;
  const int ct    = wid % 784;
  const int n     = ct / 196;
  const int pixb  = (ct % 196) * 16;
  const int col   = lane & 15;
  const int hs    = lane >> 4;
  const int pix   = pixb + col;

  const _Float16* arow = Wcat + (size_t)(mtile * 16 + col) * CIN + hs * 8;
  const _Float16* brow = xt + ((size_t)n * HW_ + pix) * CIN + hs * 8;

  v8f c = {};
#pragma unroll
  for (int k0 = 0; k0 < CIN; k0 += 32) {
    Frag16 a, b;
    a.u[0] = *(const uint4*)(arow + k0);          // K = hs*8 + 0..7
    a.u[1] = *(const uint4*)(arow + k0 + 16);     // K = hs*8 + 16..23
    b.u[0] = *(const uint4*)(brow + k0);          // K = hs*8 + 0..7
    b.u[1] = *(const uint4*)(brow + k0 + 16);     // K = hs*8 + 16..23
    c = wmma16(a.h, b.h, c);
  }

  // C layout: lane holds column pix, rows m0..m0+7 (consecutive channels).
  const int m0 = mtile * 16 + hs * 8;
  float* dst; int ch, cs;
  if (m0 < 32)       { dst = q;  ch = m0;       cs = 32;  }
  else if (m0 < 64)  { dst = kb; ch = m0 - 32;  cs = 32;  }
  else               { dst = v;  ch = m0 - 64;  cs = 256; }
  AccV cc; cc.v = c;
  float* o = dst + ((size_t)n * HW_ + pix) * cs + ch;
  *(float4*)(o)     = cc.f[0];
  *(float4*)(o + 4) = cc.f[1];
}

// ---- kernel 2: fused (q-k | pos) -> BN -> cw1 -> BN -> cw2 logits, WMMA -----
// Block = 128 threads = 4 waves; wave handles 16 pixels, loops 49 window pos.
__global__ void k2_attn_logits(const float* __restrict__ qws,
                               const float* __restrict__ kws,
                               const float* __restrict__ cpw,   // conv_p_w [2][2]
                               const float* __restrict__ bn1g, const float* __restrict__ bn1b,
                               const float* __restrict__ bn1m, const float* __restrict__ bn1v,
                               const float* __restrict__ cw1,   // [32][34]
                               const float* __restrict__ bn2g, const float* __restrict__ bn2b,
                               const float* __restrict__ bn2m, const float* __restrict__ bn2v,
                               const float* __restrict__ cw2,   // [32][32]
                               const float* __restrict__ cw2b,  // [32]
                               _Float16* __restrict__ wts) {
  __shared__ float s1[34], o1[34], s2[32], o2[32];
  {
    int t = threadIdx.x;
    if (t < 34) { float s = bn1g[t] * rsqrtf(bn1v[t] + 1e-5f); s1[t] = s; o1[t] = bn1b[t] - bn1m[t] * s; }
    if (t < 32) { float s = bn2g[t] * rsqrtf(bn2v[t] + 1e-5f); s2[t] = s; o2[t] = bn2b[t] - bn2m[t] * s; }
  }
  __syncthreads();

  const int lane = threadIdx.x & 31;
  const int wave = threadIdx.x >> 5;
  const int col  = lane & 15;
  const int hs   = lane >> 4;
  const int n    = blockIdx.x / 49;
  const int tb   = blockIdx.x % 49;
  const int pix  = tb * 64 + wave * 16 + col;
  const int ph   = pix / W_, pw = pix % W_;

  // Loop-invariant A fragments: cw1 (2 M-tiles x 2 K-tiles, K padded 34->64), cw2 (2 M-tiles).
  Frag16 a1[2][2], a2[2];
#pragma unroll
  for (int mt = 0; mt < 2; ++mt) {
    const int m = mt * 16 + col;                 // A row = lane&15
#pragma unroll
    for (int i = 0; i < 16; ++i) {
      const int kk = hs * 16 + i;   // contiguous K order
      a1[mt][0].h[i] = (_Float16)cw1[m * 34 + kk];
      const int k1i = 32 + kk;
      a1[mt][1].h[i] = (k1i < 34) ? (_Float16)cw1[m * 34 + k1i] : (_Float16)0.f;
      a2[mt].h[i]    = (_Float16)cw2[m * 32 + kk];
    }
  }
  float bias0[8], bias1[8];
#pragma unroll
  for (int r = 0; r < 8; ++r) { bias0[r] = cw2b[hs * 8 + r]; bias1[r] = cw2b[16 + hs * 8 + r]; }

  float qv[16];
  const float* qp = qws + ((size_t)n * HW_ + pix) * REL + hs * 16;
#pragma unroll
  for (int i = 0; i < 16; ++i) qv[i] = qp[i];

  const float cp00 = cpw[0], cp01 = cpw[1], cp10 = cpw[2], cp11 = cpw[3];
  const float lwp = -1.f + (2.f / 55.f) * pw, lhp = -1.f + (2.f / 55.f) * ph;

  for (int p = 0; p < KK_; ++p) {
    const int di = p / KS_ - 3, dj = p % KS_ - 3;
    const int hn = refl(ph + di, H_), wn = refl(pw + dj, W_);
    const int nbr = hn * W_ + wn;
    const float* kp = kws + ((size_t)n * HW_ + nbr) * REL + hs * 16;

    // B fragment of T: lane holds 16 channel-rows (hs*16..+15) of its pixel column.
    Frag16 b1, b2;
#pragma unroll
    for (int i = 0; i < 16; ++i) {
      const int ch = hs * 16 + i;
      const float t = (qv[i] - kp[i]) * s1[ch] + o1[ch];
      b1.h[i] = (_Float16)fmaxf(t, 0.f);
    }
    const float dlw = lwp - (-1.f + (2.f / 55.f) * wn);
    const float dlh = lhp - (-1.f + (2.f / 55.f) * hn);
    const float t32 = fmaxf((cp00 * dlw + cp01 * dlh) * s1[32] + o1[32], 0.f);
    const float t33 = fmaxf((cp10 * dlw + cp11 * dlh) * s1[33] + o1[33], 0.f);
#pragma unroll
    for (int i = 0; i < 16; ++i) b2.h[i] = (_Float16)0.f;
    b2.h[0] = (_Float16)(hs ? 0.f : t32);       // row 32 (lanes 0-15 only)
    b2.h[1] = (_Float16)(hs ? 0.f : t33);       // row 33

    v8f U0 = {}; U0 = wmma16(a1[0][0].h, b1.h, U0); U0 = wmma16(a1[0][1].h, b2.h, U0);
    v8f U1 = {}; U1 = wmma16(a1[1][0].h, b1.h, U1); U1 = wmma16(a1[1][1].h, b2.h, U1);

    // BN2 + ReLU on accumulators (channel = 8*hs + r, +16 for U1).
    float f0[8], f1[8];
#pragma unroll
    for (int r = 0; r < 8; ++r) {
      const int c0 = hs * 8 + r;
      f0[r] = fmaxf(U0[r] * s2[c0] + o2[c0], 0.f);
      f1[r] = fmaxf(U1[r] * s2[16 + c0] + o2[16 + c0], 0.f);
    }
    // C-fragment -> B-fragment reshape: half-wave swap via shfl_xor 16.
    Frag16 bt;
#pragma unroll
    for (int r = 0; r < 8; ++r) {
      const float g0 = __shfl_xor(f0[r], 16, 32);
      const float g1 = __shfl_xor(f1[r], 16, 32);
      bt.h[r]     = (_Float16)(hs ? g1 : f0[r]);   // rows 16hs + 0..7
      bt.h[8 + r] = (_Float16)(hs ? f1[r] : g0);   // rows 16hs + 8..15
    }
    v8f Wa = {}; Wa = wmma16(a2[0].h, bt.h, Wa);
    v8f Wb = {}; Wb = wmma16(a2[1].h, bt.h, Wb);

    Half8 p0, p1;
#pragma unroll
    for (int r = 0; r < 8; ++r) {
      p0.h[r] = (_Float16)(Wa[r] + bias0[r]);
      p1.h[r] = (_Float16)(Wb[r] + bias1[r]);
    }
    _Float16* wo = wts + (((size_t)(n * KK_ + p) * HW_ + pix) * 32);
    *(uint4*)(wo + hs * 8)      = p0.u;          // channels 8hs..8hs+7
    *(uint4*)(wo + 16 + hs * 8) = p1.u;          // channels 16+8hs..+7
  }
}

// ---- kernel 3: softmax over window + aggregation ----------------------------
// thread = (pixel, group g); out[n, 8g+s, pix] = sum_p softmax(w)[g,p] * v[8g+s, nbr]
__global__ void k3_softmax_agg(const _Float16* __restrict__ wts,
                               const float* __restrict__ v,
                               float* __restrict__ out) {
  const int g   = threadIdx.x & 31;
  const int pl  = threadIdx.x >> 5;
  const int n   = blockIdx.x / 392;
  const int pix = (blockIdx.x % 392) * 8 + pl;
  const int ph  = pix / W_, pw = pix % W_;

  float m = -1e30f;
  for (int p = 0; p < KK_; ++p) {
    const float wv = (float)wts[(((size_t)(n * KK_ + p) * HW_ + pix) * 32) + g];
    m = fmaxf(m, wv);
  }
  float z = 0.f;
  float acc[8];
#pragma unroll
  for (int s = 0; s < 8; ++s) acc[s] = 0.f;

  for (int p = 0; p < KK_; ++p) {
    const int di = p / KS_ - 3, dj = p % KS_ - 3;
    const int nbr = refl(ph + di, H_) * W_ + refl(pw + dj, W_);
    const float wv = (float)wts[(((size_t)(n * KK_ + p) * HW_ + pix) * 32) + g];
    const float e = expf(wv - m);
    z += e;
    const float* vp = v + ((size_t)n * HW_ + nbr) * CIN + g * 8;
#pragma unroll
    for (int s = 0; s < 8; ++s) acc[s] += e * vp[s];
  }
  const float inv = 1.f / z;
#pragma unroll
  for (int s = 0; s < 8; ++s)
    out[((size_t)n * CIN + g * 8 + s) * HW_ + pix] = acc[s] * inv;
}

extern "C" void kernel_launch(void* const* d_in, const int* in_sizes, int n_in,
                              void* d_out, int out_size, void* d_ws, size_t ws_size,
                              hipStream_t stream) {
  const float* x    = (const float*)d_in[0];
  const float* w1   = (const float*)d_in[1];
  const float* w2   = (const float*)d_in[2];
  const float* w3   = (const float*)d_in[3];
  const float* cpw  = (const float*)d_in[4];
  // d_in[5] = conv_p_b: cancels in p - unfold(p), unused.
  const float* bn1g = (const float*)d_in[6];
  const float* bn1b = (const float*)d_in[7];
  const float* bn1m = (const float*)d_in[8];
  const float* bn1v = (const float*)d_in[9];
  const float* cw1  = (const float*)d_in[10];
  const float* bn2g = (const float*)d_in[11];
  const float* bn2b = (const float*)d_in[12];
  const float* bn2m = (const float*)d_in[13];
  const float* bn2v = (const float*)d_in[14];
  const float* cw2  = (const float*)d_in[15];
  const float* cw2b = (const float*)d_in[16];
  float* out = (float*)d_out;

  // Workspace carve (all 256B-aligned): total ~62 MB.
  char* ws = (char*)d_ws;
  _Float16* Wcat = (_Float16*)(ws);                                   // 163,840 B
  _Float16* xt   = (_Float16*)(ws + 163840);                          // 6,422,528 B
  float*    qws  = (float*)(ws + 163840 + 6422528);                   // 1,605,632 B
  float*    kws  = (float*)(ws + 163840 + 6422528 + 1605632);         // 1,605,632 B
  float*    vws  = (float*)(ws + 163840 + 6422528 + 2 * 1605632);     // 12,845,056 B
  _Float16* wts  = (_Float16*)(ws + 163840 + 6422528 + 2 * 1605632 + 12845056); // 39,337,984 B

  prep_w <<<(ROWS * CIN + 255) / 256, 256, 0, stream>>>(w1, w2, w3, Wcat);
  prep_xt<<<(int)(((size_t)N_ * CIN * HW_ + 255) / 256), 256, 0, stream>>>(x, xt);
  // 20 M-tiles * 784 column-tiles = 15680 waves -> 3920 blocks of 4 waves.
  k1_gemm<<<3920, 128, 0, stream>>>(Wcat, xt, qws, kws, vws);
  // 4 batches * 49 pixel-tiles of 64 = 196 blocks of 4 waves (16 px each).
  k2_attn_logits<<<N_ * 49, 128, 0, stream>>>(qws, kws, cpw,
                                              bn1g, bn1b, bn1m, bn1v, cw1,
                                              bn2g, bn2b, bn2m, bn2v, cw2, cw2b,
                                              wts);
  // 4 * 3136 pixels / 8 per block = 1568 blocks of 256 (pixel x 32 groups).
  k3_softmax_agg<<<N_ * (HW_ / 8), 256, 0, stream>>>(wts, vws, out);
}
